// MPNN_49220325212322
// MI455X (gfx1250) — compile-verified, matrix operand order checked by reading, not run on hardware
//
#include <hip/hip_runtime.h>
#include <hip/hip_bf16.h>

typedef __attribute__((ext_vector_type(16))) _Float16 v16h;
typedef __attribute__((ext_vector_type(8)))  float    v8f;
typedef __attribute__((ext_vector_type(2)))  _Float16 h2;

#define ZZ 2
#define NNODES 2048
#define KNBR 32
#define DIM 256

// ---------------------------------------------------------------------------
// WMMA fragment load (16x32 f16 tile, row-major source with leading dim `ld`).
// Layout per CDNA5 ISA 7.12.2 (16-bit A-matrix 16x32): lanes 0-15 hold rows
// 0-15 with K={0..7,16..23}; lanes 16-31 hold rows 0-15 with K={8..15,24..31}.
// B fragments use the same loader on a column-major (transposed) staging
// buffer with ld=32, so lane&15 selects the output column.
// ---------------------------------------------------------------------------
__device__ inline v16h load_frag(const _Float16* base, int ld, int lane) {
  const int r  = lane & 15;
  const int kb = (lane & 16) ? 4 : 0;        // in h2 (2xf16) units
  const h2* p = (const h2*)(base + r * ld);
  v16h f;
#pragma unroll
  for (int i = 0; i < 4; ++i) {
    h2 t0 = p[kb + i];
    h2 t1 = p[kb + 8 + i];
    f[2 * i]     = t0.x; f[2 * i + 1] = t0.y;
    f[2 * i + 8] = t1.x; f[2 * i + 9] = t1.y;
  }
  return f;
}

// ---------------------------------------------------------------------------
// Block GEMM: out[32xN] = act(A[32xK] @ W[KxN] + bias).  8 waves, each wave
// owns N/128 column tiles x 2 row tiles.  W chunks (32 x N) are staged into
// LDS transposed so B fragments are contiguous along K.  f32 accumulate via
// v_wmma_f32_16x16x32_f16.
// ---------------------------------------------------------------------------
template<int K, int N>
__device__ inline void gemm32(const _Float16* __restrict__ A,
                              const _Float16* __restrict__ W,
                              const float* __restrict__ bias, bool act,
                              _Float16* __restrict__ out,
                              _Float16* __restrict__ bstage) {
  constexpr int NT = N / 128;        // column tiles per wave (8 waves)
  constexpr int NH = N / 2;
  const int tid  = threadIdx.x;
  const int lane = tid & 31;
  const int wave = tid >> 5;

  const v8f vzero = {};
  v8f acc[2][NT];
#pragma unroll
  for (int m = 0; m < 2; ++m)
#pragma unroll
    for (int t = 0; t < NT; ++t) acc[m][t] = vzero;

  const int nk = K >> 5;
  for (int kc = 0; kc < nk; ++kc) {
    __syncthreads();
    // stage W chunk [32 x N] transposed -> bstage[col*32 + k]
    const _Float16* wp = W + kc * 32 * N;
    for (int e2 = tid; e2 < 16 * N; e2 += 256) {
      int k  = e2 / NH;              // constexpr power-of-two -> shift
      int c2 = e2 - k * NH;
      h2 v = ((const h2*)wp)[e2];
      int c = c2 * 2;
      bstage[c * 32 + k]       = v.x;
      bstage[(c + 1) * 32 + k] = v.y;
    }
    if (kc + 1 < nk)   // pull next weight chunk toward L2/L0 (global_prefetch)
      __builtin_prefetch((const void*)(W + (kc + 1) * 32 * N + tid * 8), 0, 1);
    __syncthreads();

    v16h a0 = load_frag(A + kc * 32, K, lane);
    v16h a1 = load_frag(A + 16 * K + kc * 32, K, lane);
#pragma unroll
    for (int t = 0; t < NT; ++t) {
      const int col0 = (wave * NT + t) << 4;
      v16h b = load_frag(bstage + col0 * 32, 32, lane);
      acc[0][t] = __builtin_amdgcn_wmma_f32_16x16x32_f16(
          false, a0, false, b, (short)0, acc[0][t], false, false);
      acc[1][t] = __builtin_amdgcn_wmma_f32_16x16x32_f16(
          false, a1, false, b, (short)0, acc[1][t], false, false);
    }
  }

  // epilogue: C/D layout -> row = vgpr + 8*(lane>=16), col = lane&15
  const int cl = lane & 15;
  const int rh = (lane >> 4) << 3;
#pragma unroll
  for (int t = 0; t < NT; ++t) {
    const int col = ((wave * NT + t) << 4) + cl;
    const float bi = bias[col];
#pragma unroll
    for (int m = 0; m < 2; ++m) {
      const int rbase = (m << 4) + rh;
#pragma unroll
      for (int r = 0; r < 8; ++r) {
        float v = acc[m][t][r] + bi;
        if (act) v = v / (1.f + __expf(-v));   // SiLU
        out[(rbase + r) * N + col] = (_Float16)v;
      }
    }
  }
  __syncthreads();
}

// ---------------------------------------------------------------------------
// LayerNorm over 32 rows x 256 cols in LDS (in place). 8 threads per row.
// ---------------------------------------------------------------------------
__device__ inline void ln_rows(float* __restrict__ x,
                               const float* __restrict__ g,
                               const float* __restrict__ b,
                               float* __restrict__ red, int tid) {
  const int row = tid >> 3;
  const int sub = tid & 7;
  const int c0  = sub * 32;
  float s = 0.f, s2 = 0.f;
#pragma unroll
  for (int c = c0; c < c0 + 32; ++c) {
    float v = x[row * 256 + c];
    s += v; s2 += v * v;
  }
  red[row * 16 + sub]     = s;
  red[row * 16 + 8 + sub] = s2;
  __syncthreads();
  float ts = 0.f, ts2 = 0.f;
#pragma unroll
  for (int i = 0; i < 8; ++i) {
    ts  += red[row * 16 + i];
    ts2 += red[row * 16 + 8 + i];
  }
  const float mean = ts * (1.f / 256.f);
  const float var  = ts2 * (1.f / 256.f) - mean * mean;
  const float rstd = rsqrtf(var + 1e-5f);
#pragma unroll
  for (int c = c0; c < c0 + 32; ++c) {
    float v = (x[row * 256 + c] - mean) * rstd;
    x[row * 256 + c] = v * g[c] + b[c];
  }
}

// ---------------------------------------------------------------------------
// Kernel 0: f32 -> f16 weight conversion
// ---------------------------------------------------------------------------
__global__ void cvt_kernel(const float* __restrict__ s,
                           _Float16* __restrict__ d, int n) {
  int i = blockIdx.x * blockDim.x + threadIdx.x;
  if (i < n) d[i] = (_Float16)s[i];
}

// ---------------------------------------------------------------------------
// Kernel 1: node message MLP + masked aggregation.  One block per (z,n).
// x = nodes + sum_k mask * MLP([node_i | node_j | edge])  -> xbuf
// ---------------------------------------------------------------------------
__global__ __launch_bounds__(256) void node_msg_kernel(
    const float* __restrict__ nodes, const float* __restrict__ edges,
    const int* __restrict__ nbrs, const float* __restrict__ mask,
    const _Float16* w0, const float* b0, const _Float16* w1, const float* b1,
    const _Float16* w2, const float* b2, const _Float16* w3, const float* b3,
    float* __restrict__ xbuf) {
  extern __shared__ char smem[];
  _Float16* msg = (_Float16*)smem;          // 32 x 768
  _Float16* h0  = msg + 32 * 768;           // 32 x 256
  _Float16* h1  = h0 + 32 * 256;            // 32 x 256
  _Float16* bst = h1 + 32 * 256;            // 256 x 32 (transposed W chunk)
  float* smask  = (float*)(bst + 256 * 32); // 32
  int* snbr     = (int*)(smask + 32);       // 32

  const int tid = threadIdx.x;
  const int zn  = blockIdx.x;               // z*N + n
  const int z   = zn >> 11;
  const int nodeBase = zn * 256;
  const int edgeBase = zn * 32 * 256;

  if (tid < 32) {
    snbr[tid]  = nbrs[zn * 32 + tid];
    smask[tid] = mask[zn * 32 + tid];
  }
  __syncthreads();
  {
    _Float16 hv = (_Float16)nodes[nodeBase + tid];   // node_i broadcast
#pragma unroll
    for (int k = 0; k < 32; ++k) msg[k * 768 + tid] = hv;
  }
  for (int e = tid; e < 32 * 256; e += 256) {
    int k = e >> 8, c = e & 255;
    msg[k * 768 + 256 + c] =
        (_Float16)nodes[(z * NNODES + snbr[k]) * 256 + c];   // gathered node_j
    msg[k * 768 + 512 + c] = (_Float16)edges[edgeBase + e];  // edge feature
  }
  __syncthreads();

  gemm32<768, 256>(msg, w0, b0, true,  h0, bst);
  gemm32<256, 256>(h0,  w1, b1, true,  h1, bst);
  gemm32<256, 256>(h1,  w2, b2, true,  h0, bst);
  gemm32<256, 256>(h0,  w3, b3, false, h1, bst);

  float a = 0.f;
#pragma unroll
  for (int k = 0; k < 32; ++k) a += (float)h1[k * 256 + tid] * smask[k];
  xbuf[nodeBase + tid] = nodes[nodeBase + tid] + a;
}

// ---------------------------------------------------------------------------
// Kernel 2: LN1 -> FFN (256->1024->256, SiLU) -> residual -> LN2 -> nodes_out
// One block per 32 nodes.
// ---------------------------------------------------------------------------
__global__ __launch_bounds__(256) void node_update_kernel(
    const float* __restrict__ xbuf,
    const float* __restrict__ g1, const float* __restrict__ bb1,
    const _Float16* fw0, const float* fb0,
    const _Float16* fw1, const float* fb1,
    const float* __restrict__ g2, const float* __restrict__ bb2,
    float* __restrict__ nodes_out) {
  extern __shared__ char smem[];
  float* xf      = (float*)smem;                 // 32 x 256 f32
  _Float16* a16  = (_Float16*)(xf + 32 * 256);   // 32 x 256
  _Float16* hbig = a16 + 32 * 256;               // 32 x 1024
  _Float16* o16  = hbig + 32 * 1024;             // 32 x 256
  _Float16* bst  = o16 + 32 * 256;               // 1024 x 32
  float* red     = (float*)(bst + 32 * 1024);    // 32 x 16

  const int tid  = threadIdx.x;
  const int base = blockIdx.x * 32 * 256;

  for (int e = tid; e < 8192; e += 256) xf[e] = xbuf[base + e];
  __syncthreads();
  ln_rows(xf, g1, bb1, red, tid);                // xf = n1
  __syncthreads();
  for (int e = tid; e < 8192; e += 256) a16[e] = (_Float16)xf[e];
  __syncthreads();

  gemm32<256, 1024>(a16,  fw0, fb0, true,  hbig, bst);
  gemm32<1024, 256>(hbig, fw1, fb1, false, o16,  bst);

  for (int e = tid; e < 8192; e += 256) xf[e] += (float)o16[e];  // ff + n1
  __syncthreads();
  ln_rows(xf, g2, bb2, red, tid);
  __syncthreads();
  for (int e = tid; e < 8192; e += 256) nodes_out[base + e] = xf[e];
}

// ---------------------------------------------------------------------------
// Kernel 3: edge MLP on updated nodes + residual + per-row LN -> edges_out.
// One block per (z,n) (32 edge rows).
// ---------------------------------------------------------------------------
__global__ __launch_bounds__(256) void edge_kernel(
    const float* __restrict__ nodes_up, const float* __restrict__ edges,
    const int* __restrict__ nbrs,
    const _Float16* w0, const float* b0, const _Float16* w1, const float* b1,
    const _Float16* w2, const float* b2, const _Float16* w3, const float* b3,
    const float* __restrict__ elng, const float* __restrict__ elnb,
    float* __restrict__ edges_out) {
  extern __shared__ char smem[];
  _Float16* msg = (_Float16*)smem;          // 32 x 768
  _Float16* h0  = msg + 32 * 768;
  _Float16* h1  = h0 + 32 * 256;
  _Float16* bst = h1 + 32 * 256;            // 256 x 32
  float* ef     = (float*)(bst + 256 * 32); // 32 x 256 f32
  float* red    = ef + 32 * 256;            // 32 x 16
  int* snbr     = (int*)(red + 32 * 16);    // 32

  const int tid = threadIdx.x;
  const int zn  = blockIdx.x;
  const int z   = zn >> 11;
  const int nodeBase = zn * 256;
  const int edgeBase = zn * 32 * 256;

  if (tid < 32) snbr[tid] = nbrs[zn * 32 + tid];
  __syncthreads();
  {
    _Float16 hv = (_Float16)nodes_up[nodeBase + tid];
#pragma unroll
    for (int k = 0; k < 32; ++k) msg[k * 768 + tid] = hv;
  }
  for (int e = tid; e < 32 * 256; e += 256) {
    int k = e >> 8, c = e & 255;
    msg[k * 768 + 256 + c] =
        (_Float16)nodes_up[(z * NNODES + snbr[k]) * 256 + c];
    msg[k * 768 + 512 + c] = (_Float16)edges[edgeBase + e];
  }
  __syncthreads();

  gemm32<768, 256>(msg, w0, b0, true,  h0, bst);
  gemm32<256, 256>(h0,  w1, b1, true,  h1, bst);
  gemm32<256, 256>(h1,  w2, b2, true,  h0, bst);
  gemm32<256, 256>(h0,  w3, b3, false, h1, bst);

  for (int e = tid; e < 32 * 256; e += 256)
    ef[e] = edges[edgeBase + e] + (float)h1[e];
  __syncthreads();
  ln_rows(ef, elng, elnb, red, tid);
  __syncthreads();
  for (int e = tid; e < 32 * 256; e += 256)
    edges_out[edgeBase + e] = ef[e];
}

// ---------------------------------------------------------------------------
// Host launcher
// ---------------------------------------------------------------------------
extern "C" void kernel_launch(void* const* d_in, const int* in_sizes, int n_in,
                              void* d_out, int out_size, void* d_ws,
                              size_t ws_size, hipStream_t stream) {
  (void)in_sizes; (void)n_in; (void)out_size; (void)ws_size;
  const float* nodes = (const float*)d_in[0];
  const float* edges = (const float*)d_in[1];
  const int*   nbrs  = (const int*)d_in[2];
  const float* mask  = (const float*)d_in[3];
  const float* nm_w0 = (const float*)d_in[4];  const float* nm_b0 = (const float*)d_in[5];
  const float* nm_w1 = (const float*)d_in[6];  const float* nm_b1 = (const float*)d_in[7];
  const float* nm_w2 = (const float*)d_in[8];  const float* nm_b2 = (const float*)d_in[9];
  const float* nm_w3 = (const float*)d_in[10]; const float* nm_b3 = (const float*)d_in[11];
  const float* ln1_g = (const float*)d_in[12]; const float* ln1_b = (const float*)d_in[13];
  const float* ffn_w0 = (const float*)d_in[14]; const float* ffn_b0 = (const float*)d_in[15];
  const float* ffn_w1 = (const float*)d_in[16]; const float* ffn_b1 = (const float*)d_in[17];
  const float* ln2_g = (const float*)d_in[18]; const float* ln2_b = (const float*)d_in[19];
  const float* em_w0 = (const float*)d_in[20]; const float* em_b0 = (const float*)d_in[21];
  const float* em_w1 = (const float*)d_in[22]; const float* em_b1 = (const float*)d_in[23];
  const float* em_w2 = (const float*)d_in[24]; const float* em_b2 = (const float*)d_in[25];
  const float* em_w3 = (const float*)d_in[26]; const float* em_b3 = (const float*)d_in[27];
  const float* eln_g = (const float*)d_in[28]; const float* eln_b = (const float*)d_in[29];

  float* nodes_out = (float*)d_out;
  float* edges_out = nodes_out + (size_t)ZZ * NNODES * DIM;

  // workspace carve-out (256B aligned)
  char* ws = (char*)d_ws;
  size_t off = 0;
  auto carve = [&](size_t bytes) -> void* {
    void* p = ws + off;
    off += (bytes + 255) & ~(size_t)255;
    return p;
  };
  _Float16* nm0h = (_Float16*)carve(768 * 256 * 2);
  _Float16* nm1h = (_Float16*)carve(256 * 256 * 2);
  _Float16* nm2h = (_Float16*)carve(256 * 256 * 2);
  _Float16* nm3h = (_Float16*)carve(256 * 256 * 2);
  _Float16* f0h  = (_Float16*)carve(256 * 1024 * 2);
  _Float16* f1h  = (_Float16*)carve(1024 * 256 * 2);
  _Float16* e0h  = (_Float16*)carve(768 * 256 * 2);
  _Float16* e1h  = (_Float16*)carve(256 * 256 * 2);
  _Float16* e2h  = (_Float16*)carve(256 * 256 * 2);
  _Float16* e3h  = (_Float16*)carve(256 * 256 * 2);
  float* xbuf    = (float*)carve((size_t)ZZ * NNODES * DIM * 4);

  auto cvt = [&](const float* s, _Float16* d, int n) {
    cvt_kernel<<<(n + 255) / 256, 256, 0, stream>>>(s, d, n);
  };
  cvt(nm_w0, nm0h, 768 * 256);
  cvt(nm_w1, nm1h, 256 * 256);
  cvt(nm_w2, nm2h, 256 * 256);
  cvt(nm_w3, nm3h, 256 * 256);
  cvt(ffn_w0, f0h, 256 * 1024);
  cvt(ffn_w1, f1h, 1024 * 256);
  cvt(em_w0, e0h, 768 * 256);
  cvt(em_w1, e1h, 256 * 256);
  cvt(em_w2, e2h, 256 * 256);
  cvt(em_w3, e3h, 256 * 256);

  constexpr int SMEM1 = (32 * 768 + 32 * 256 * 2 + 256 * 32) * 2 + 32 * 4 + 32 * 4;
  constexpr int SMEM2 = 32 * 256 * 4 + (32 * 256 + 32 * 1024 + 32 * 256 + 1024 * 32) * 2
                        + 32 * 16 * 4;
  constexpr int SMEM3 = (32 * 768 + 32 * 256 * 2 + 256 * 32) * 2 + 32 * 256 * 4
                        + 32 * 16 * 4 + 32 * 4;

  (void)hipFuncSetAttribute((const void*)node_msg_kernel,
                            hipFuncAttributeMaxDynamicSharedMemorySize, SMEM1);
  (void)hipFuncSetAttribute((const void*)node_update_kernel,
                            hipFuncAttributeMaxDynamicSharedMemorySize, SMEM2);
  (void)hipFuncSetAttribute((const void*)edge_kernel,
                            hipFuncAttributeMaxDynamicSharedMemorySize, SMEM3);

  node_msg_kernel<<<ZZ * NNODES, 256, SMEM1, stream>>>(
      nodes, edges, nbrs, mask,
      nm0h, nm_b0, nm1h, nm_b1, nm2h, nm_b2, nm3h, nm_b3, xbuf);

  node_update_kernel<<<ZZ * NNODES / 32, 256, SMEM2, stream>>>(
      xbuf, ln1_g, ln1_b, f0h, ffn_b0, f1h, ffn_b1, ln2_g, ln2_b, nodes_out);

  edge_kernel<<<ZZ * NNODES, 256, SMEM3, stream>>>(
      nodes_out, edges, nbrs,
      e0h, em_b0, e1h, em_b1, e2h, em_b2, e3h, em_b3,
      eln_g, eln_b, edges_out);
}
